// Decoder_49005576847865
// MI455X (gfx1250) — compile-verified, matrix operand order checked by reading, not run on hardware
//
#include <hip/hip_runtime.h>
#include <hip/hip_bf16.h>
#include <stdint.h>

// ---------------------------------------------------------------------------
// CDNA5 (gfx1250) decoder: f16 WMMA GEMMs + cached cross-attention K/V.
// ---------------------------------------------------------------------------

typedef __attribute__((ext_vector_type(16))) _Float16 v16h;
typedef __attribute__((ext_vector_type(8)))  _Float16 v8h;
typedef __attribute__((ext_vector_type(8)))  float    v8f;

#define B_    16
#define LENC  2048
#define FEAT  128
#define DMOD  512
#define DFF   2048
#define NL    4
#define TS    256
#define NT    16
#define NH    8
#define DH    64

// ------------------------------- conversion --------------------------------
__global__ void k_f32_to_f16(const float* __restrict__ src,
                             _Float16* __restrict__ dst, int n) {
  int i = blockIdx.x * blockDim.x + threadIdx.x;
  if (i < n) dst[i] = (_Float16)src[i];
}

// padding mask from encoder_input: mask[b*LENC+j] = 1 if all FEAT values == 0
__global__ void k_pad_mask(const float* __restrict__ enc_in,
                           float* __restrict__ mask) {
  int j = blockIdx.x * blockDim.x + threadIdx.x;
  if (j >= B_ * LENC) return;
  const float* p = enc_in + (size_t)j * FEAT;
  float allz = 1.f;
  for (int f = 0; f < FEAT; ++f) {
    if (p[f] != 0.f) { allz = 0.f; break; }
  }
  mask[j] = allz;
}

// ------------------------------- WMMA GEMM ---------------------------------
// C[M,N] (+bias, opt relu) = A_f16[M,K] * B_f16[K,N].  Block = 128 thr = 4
// waves; each wave owns one 16x16 tile; block tile = 16 x 64.  K step = 64
// (2 WMMAs per barrier pair).  A staged row-major [16][64]; B staged
// column-major [64 cols][64 kk] with 72-f16 (144B) padded rows so every
// 8-element fragment chunk is contiguous + 16B aligned -> ds_load_b128.
// flags: bit0 = relu, bit1 = store f16 (else f32).
#define BKPAD 72

static __device__ __forceinline__ v16h frag16(const _Float16* p0,
                                              const _Float16* p1) {
  v8h lo = *(const v8h*)p0;
  v8h hi = *(const v8h*)p1;
  return __builtin_shufflevector(lo, hi, 0, 1, 2, 3, 4, 5, 6, 7,
                                 8, 9, 10, 11, 12, 13, 14, 15);
}

__global__ void k_gemm_f16_wmma(const _Float16* __restrict__ A, int lda,
                                const _Float16* __restrict__ Bm, int ldb,
                                const float* __restrict__ bias,
                                void* __restrict__ C, int ldc,
                                int K, int flags) {
  __shared__ _Float16 sA[16 * 64];         // [row][kk]
  __shared__ _Float16 sBt[64 * BKPAD];     // [col][kk], padded
  const int tid  = threadIdx.x;
  const int lane = tid & 31;
  const int wave = tid >> 5;         // 0..3 -> N sub-tile
  const int half = lane >> 4;        // 0/1  (wave32 fragment halves)
  const int mrow = lane & 15;        // A row / B col within tile
  const int m0 = blockIdx.x * 16;
  const int n0 = blockIdx.y * 64;
  const int ncol = (wave << 4) + mrow;
  const int hb = half << 3;          // half * 8

  v8f c = {};
  for (int k0 = 0; k0 < K; k0 += 64) {
    {  // stage A tile 16x64: one 16B chunk per thread
      int idx = tid << 3;            // 0..1016
      int r = idx >> 6, kkb = idx & 63;
      *(v8h*)&sA[idx] = *(const v8h*)(A + (size_t)(m0 + r) * lda + k0 + kkb);
    }
#pragma unroll
    for (int u = 0; u < 4; ++u) {    // stage B tile 64x64, transposing
      int idx = (tid + u * 128) << 3;
      int kk = idx >> 6;             // 0..63
      int c0 = idx & 63;             // multiple of 8
      v8h vv = *(const v8h*)(Bm + (size_t)(k0 + kk) * ldb + n0 + c0);
#pragma unroll
      for (int j = 0; j < 8; ++j) sBt[(c0 + j) * BKPAD + kk] = vv[j];
    }
    __syncthreads();

    // wave32 16-bit fragment layout (ISA 7.12.2):
    // elems 0..7 -> K = half*8 + i ; elems 8..15 -> K = 16 + half*8 + i
    const _Float16* pa = &sA[(mrow << 6) + hb];
    const _Float16* pb = &sBt[ncol * BKPAD + hb];
    v16h a0 = frag16(pa,      pa + 16);
    v16h b0 = frag16(pb,      pb + 16);
    c = __builtin_amdgcn_wmma_f32_16x16x32_f16(
        false, a0, false, b0, (short)0, c, false, false);
    v16h a1 = frag16(pa + 32, pa + 48);
    v16h b1 = frag16(pb + 32, pb + 48);
    c = __builtin_amdgcn_wmma_f32_16x16x32_f16(
        false, a1, false, b1, (short)0, c, false, false);
    __syncthreads();
  }

  const int gcol = n0 + ncol;
  const float bv = bias ? bias[gcol] : 0.f;
#pragma unroll
  for (int r = 0; r < 8; ++r) {
    int grow = m0 + r + (half << 3);   // C layout: VGPR r -> row r + 8*half
    float v = c[r] + bv;
    if (flags & 1) v = v > 0.f ? v : 0.f;
    if (flags & 2) ((_Float16*)C)[(size_t)grow * ldc + gcol] = (_Float16)v;
    else           ((float*)C)[(size_t)grow * ldc + gcol] = v;
  }
}

// ------------------------------- embedding ---------------------------------
// x[b,d] = (tok @ emb_W + emb_b) * sqrt(d) + pos_enc(t, d);  tok = ones at t=0
__global__ void k_embed(const float* __restrict__ out_tokens,  // d_out base
                        const _Float16* __restrict__ embW,
                        const float* __restrict__ embB,
                        float* __restrict__ x, _Float16* __restrict__ xh,
                        int t) {
  int b = blockIdx.x, dcol = threadIdx.x;     // 512 threads
  const float* tp = (t == 0) ? nullptr
      : out_tokens + (size_t)b * (NT * TS) + (size_t)(t - 1) * TS;
  float acc = 0.f;
  for (int k = 0; k < TS; ++k) {
    float tv = tp ? tp[k] : 1.0f;
    acc += tv * (float)embW[k * DMOD + dcol];
  }
  acc = (acc + embB[dcol]) * 22.62741699796952f;  // sqrt(512)
  float factor = __powf(10000.f, -(float)(2 * (dcol / 2)) / (float)DMOD);
  float ang = (float)t * factor;
  acc += (dcol & 1) ? __cosf(ang) : __sinf(ang);
  x[b * DMOD + dcol]  = acc;
  xh[b * DMOD + dcol] = (_Float16)acc;
}

// ------------------------- incremental self-attn ---------------------------
// grid (B, H), block 64.  Causal attention of the new token over cached K/V.
__global__ void k_self_attn(const float* __restrict__ q,
                            const float* __restrict__ kc,   // [B,NT,DMOD]
                            const float* __restrict__ vc,
                            float* __restrict__ o, _Float16* __restrict__ oh,
                            int t) {
  int b = blockIdx.x, h = blockIdx.y, tid = threadIdx.x;
  __shared__ float qv[DH];
  __shared__ float sl[NT];
  qv[tid] = q[b * DMOD + h * DH + tid];
  __syncthreads();
  if (tid <= t) {
    const float* kp = kc + ((size_t)b * NT + tid) * DMOD + h * DH;
    float dot = 0.f;
    for (int d = 0; d < DH; ++d) dot += qv[d] * kp[d];
    sl[tid] = dot * 0.125f;  // 1/sqrt(64)
  }
  __syncthreads();
  float mx = -1e30f;
  for (int j = 0; j <= t; ++j) mx = fmaxf(mx, sl[j]);
  float sum = 0.f;
  for (int j = 0; j <= t; ++j) sum += __expf(sl[j] - mx);
  float inv = 1.f / sum;
  float od = 0.f;
  for (int j = 0; j <= t; ++j) {
    float w = __expf(sl[j] - mx) * inv;
    od += w * vc[((size_t)b * NT + j) * DMOD + h * DH + tid];
  }
  o[b * DMOD + h * DH + tid]  = od;
  oh[b * DMOD + h * DH + tid] = (_Float16)od;
}

// ----------------------------- cross-attn ----------------------------------
// grid (B, H), block 256.  Streams precomputed f16 K/V (the bandwidth-heavy
// part: ~268MB per step @ 23.3 TB/s).
__global__ void k_cross_attn(const float* __restrict__ q,
                             const _Float16* __restrict__ K,  // [B,LENC,DMOD]
                             const _Float16* __restrict__ V,
                             const float* __restrict__ mask,  // [B,LENC]
                             float* __restrict__ o, _Float16* __restrict__ oh) {
  int b = blockIdx.x, h = blockIdx.y, tid = threadIdx.x;
  __shared__ float qv[DH];
  __shared__ float logits[LENC];
  __shared__ float red[256];
  if (tid < DH) qv[tid] = q[b * DMOD + h * DH + tid];
  __syncthreads();

  float lmax = -1e30f;
  for (int j = tid; j < LENC; j += 256) {
    const _Float16* kp = K + ((size_t)b * LENC + j) * DMOD + h * DH;
    __builtin_prefetch(kp + 256 * DMOD, 0, 1);   // global_prefetch_b8
    float dot = 0.f;
#pragma unroll 8
    for (int d = 0; d < DH; ++d) dot += qv[d] * (float)kp[d];
    dot *= 0.125f;
    if (mask[b * LENC + j] > 0.f) dot = -3.0e38f;
    logits[j] = dot;
    lmax = fmaxf(lmax, dot);
  }
  red[tid] = lmax; __syncthreads();
  for (int s = 128; s > 0; s >>= 1) {
    if (tid < s) red[tid] = fmaxf(red[tid], red[tid + s]);
    __syncthreads();
  }
  lmax = red[0]; __syncthreads();

  float lsum = 0.f;
  for (int j = tid; j < LENC; j += 256) {
    float e = __expf(logits[j] - lmax);
    logits[j] = e; lsum += e;
  }
  red[tid] = lsum; __syncthreads();
  for (int s = 128; s > 0; s >>= 1) {
    if (tid < s) red[tid] += red[tid + s];
    __syncthreads();
  }
  float inv = 1.f / red[0];
  __syncthreads();

  int g = tid >> 6, d = tid & 63;
  float acc = 0.f;
  for (int j = g * (LENC / 4); j < (g + 1) * (LENC / 4); ++j)
    acc += logits[j] * (float)V[((size_t)b * LENC + j) * DMOD + h * DH + d];
  red[tid] = acc; __syncthreads();
  if (tid < DH) {
    float s = (red[tid] + red[tid + 64] + red[tid + 128] + red[tid + 192]) * inv;
    o[b * DMOD + h * DH + tid]  = s;
    oh[b * DMOD + h * DH + tid] = (_Float16)s;
  }
}

// ------------------------ residual + layernorm -----------------------------
__global__ void k_ln(const float* __restrict__ xin, const float* __restrict__ res,
                     const float* __restrict__ g, const float* __restrict__ bb,
                     float* __restrict__ xout, _Float16* __restrict__ xh) {
  int b = blockIdx.x, tid = threadIdx.x;   // 512 threads
  __shared__ float red[DMOD];
  float v = xin[b * DMOD + tid] + res[b * DMOD + tid];
  red[tid] = v; __syncthreads();
  for (int s = 256; s > 0; s >>= 1) {
    if (tid < s) red[tid] += red[tid + s];
    __syncthreads();
  }
  float mean = red[0] * (1.f / DMOD); __syncthreads();
  float dv = v - mean;
  red[tid] = dv * dv; __syncthreads();
  for (int s = 256; s > 0; s >>= 1) {
    if (tid < s) red[tid] += red[tid + s];
    __syncthreads();
  }
  float var = red[0] * (1.f / DMOD);
  float y = dv * rsqrtf(var + 1e-6f) * g[tid] + bb[tid];
  xout[b * DMOD + tid] = y;
  xh[b * DMOD + tid]   = (_Float16)y;
}

// ------------------------- TDM warm-up (async tensor) ----------------------
#if __has_builtin(__builtin_amdgcn_tensor_load_to_lds) && \
    __has_builtin(__builtin_amdgcn_s_wait_tensorcnt)
#define HAVE_TDM 1
typedef __attribute__((ext_vector_type(4))) unsigned int v4u;
typedef __attribute__((ext_vector_type(8))) int v8i;
typedef __attribute__((ext_vector_type(4))) int v4i;

// Pulls a 16x64-element f16 tile of the converted encoder activations into
// LDS via the Tensor Data Mover (warms GL2 for the cross-K/V GEMMs).
// This toolchain exposes the 6-arg builtin:
//   (uint32x4 g0, int32x8 g1, int32x4 g2, int32x4 g3, int32x8 g4, i32 cpol)
__global__ void k_tdm_warm(const _Float16* __restrict__ src,
                           float* __restrict__ sink) {
  __shared__ _Float16 tile[2048];
  tile[threadIdx.x] = (_Float16)0;
  unsigned long long ga = (unsigned long long)(uintptr_t)src;
  // D# group0: count=1 | lds_addr=0 | global_addr | type=2 ("image")
  v4u g0 = { 1u, 0u, (unsigned int)(ga & 0xffffffffu),
             (unsigned int)(((ga >> 32) & 0x01ffffffu) | 0x80000000u) };
  // D# group1: data_size=2B, tensor_dim0=512, tensor_dim1=2048,
  //            tile 64x16, tensor_dim0_stride=512, tensor_dim1_stride=512*2048
  v8i g1 = { 0x00010000, (int)(512u << 16), (int)(2048u << 16),
             (int)(64u << 16), 16, 512, 0, 16 };
  v4i g2 = { 0, 0, 0, 0 };
  v4i g3 = { 0, 0, 0, 0 };
  v8i g4 = { 0, 0, 0, 0, 0, 0, 0, 0 };
  __builtin_amdgcn_tensor_load_to_lds(g0, g1, g2, g3, g4, 0);
  __builtin_amdgcn_s_wait_tensorcnt(0);
  __syncthreads();
  if (threadIdx.x == 0) sink[0] = (float)tile[0];
}
#endif

// =============================== host side =================================
static inline void gemm(hipStream_t s, const _Float16* A, int lda,
                        const _Float16* Bm, int ldb, const float* bias,
                        void* C, int ldc, int M, int N, int K, int flags) {
  dim3 grid(M / 16, N / 64);
  k_gemm_f16_wmma<<<grid, 128, 0, s>>>(A, lda, Bm, ldb, bias, C, ldc, K, flags);
}

extern "C" void kernel_launch(void* const* d_in, const int* in_sizes, int n_in,
                              void* d_out, int out_size, void* d_ws, size_t ws_size,
                              hipStream_t stream) {
  (void)in_sizes; (void)n_in; (void)out_size; (void)ws_size;
  const float* encoder_output = (const float*)d_in[0];
  const float* encoder_input  = (const float*)d_in[1];
  const float* emb_W   = (const float*)d_in[2];
  const float* emb_b   = (const float*)d_in[3];
  const float* out_W   = (const float*)d_in[4];
  const float* out_b   = (const float*)d_in[5];
  const float* self_W  = (const float*)d_in[6];
  const float* self_b  = (const float*)d_in[7];
  const float* cross_W = (const float*)d_in[8];
  const float* cross_b = (const float*)d_in[9];
  const float* ffn_W1  = (const float*)d_in[10];
  const float* ffn_b1  = (const float*)d_in[11];
  const float* ffn_W2  = (const float*)d_in[12];
  const float* ffn_b2  = (const float*)d_in[13];
  const float* ln_g    = (const float*)d_in[14];
  const float* ln_b    = (const float*)d_in[15];

  // ---- workspace layout (running offsets, 256B aligned) ----
  char* ws = (char*)d_ws;
  size_t off = 0;
  auto alloc = [&](size_t bytes) -> char* {
    char* p = ws + off;
    off = (off + bytes + 255) & ~(size_t)255;
    return p;
  };
  _Float16* enc_h    = (_Float16*)alloc((size_t)B_ * LENC * DMOD * 2);
  _Float16* selfW_h  = (_Float16*)alloc((size_t)NL * 4 * DMOD * DMOD * 2);
  _Float16* crossW_h = (_Float16*)alloc((size_t)NL * 4 * DMOD * DMOD * 2);
  _Float16* ffn1_h   = (_Float16*)alloc((size_t)NL * DMOD * DFF * 2);
  _Float16* ffn2_h   = (_Float16*)alloc((size_t)NL * DFF * DMOD * 2);
  _Float16* embW_h   = (_Float16*)alloc((size_t)TS * DMOD * 2);
  _Float16* outW_h   = (_Float16*)alloc((size_t)DMOD * TS * 2);
  _Float16* Kc_h     = (_Float16*)alloc((size_t)NL * B_ * LENC * DMOD * 2);
  _Float16* Vc_h     = (_Float16*)alloc((size_t)NL * B_ * LENC * DMOD * 2);
  float*    padmask  = (float*)alloc((size_t)B_ * LENC * 4);
  float*    kcache   = (float*)alloc((size_t)NL * B_ * NT * DMOD * 4);
  float*    vcache   = (float*)alloc((size_t)NL * B_ * NT * DMOD * 4);
  float*    xbuf     = (float*)alloc((size_t)B_ * DMOD * 4);
  _Float16* xh       = (_Float16*)alloc((size_t)B_ * DMOD * 2);
  float*    qbuf     = (float*)alloc((size_t)B_ * DMOD * 4);
  float*    obuf     = (float*)alloc((size_t)B_ * DMOD * 4);
  _Float16* oh       = (_Float16*)alloc((size_t)B_ * DMOD * 2);
  float*    abuf     = (float*)alloc((size_t)B_ * DMOD * 4);
  _Float16* hbuf_h   = (_Float16*)alloc((size_t)B_ * DFF * 2);
  float*    sink     = (float*)alloc(256);

  auto conv = [&](const float* s, _Float16* d, int n) {
    k_f32_to_f16<<<(n + 255) / 256, 256, 0, stream>>>(s, d, n);
  };

  // ---- one-time (per call) setup: f16 weights, pad mask, cross K/V ----
  conv(encoder_output, enc_h, B_ * LENC * DMOD);
  conv(self_W,  selfW_h,  NL * 4 * DMOD * DMOD);
  conv(cross_W, crossW_h, NL * 4 * DMOD * DMOD);
  conv(ffn_W1,  ffn1_h,   NL * DMOD * DFF);
  conv(ffn_W2,  ffn2_h,   NL * DFF * DMOD);
  conv(emb_W,   embW_h,   TS * DMOD);
  conv(out_W,   outW_h,   DMOD * TS);
  k_pad_mask<<<(B_ * LENC + 255) / 256, 256, 0, stream>>>(encoder_input, padmask);
#if defined(HAVE_TDM)
  k_tdm_warm<<<1, 32, 0, stream>>>(enc_h, sink);
#else
  (void)sink;
#endif
  for (int l = 0; l < NL; ++l) {
    gemm(stream, enc_h, DMOD, crossW_h + (size_t)(l * 4 + 1) * DMOD * DMOD, DMOD,
         cross_b + (size_t)(l * 4 + 1) * DMOD,
         Kc_h + (size_t)l * B_ * LENC * DMOD, DMOD,
         B_ * LENC, DMOD, DMOD, /*f16 out*/ 2);
    gemm(stream, enc_h, DMOD, crossW_h + (size_t)(l * 4 + 2) * DMOD * DMOD, DMOD,
         cross_b + (size_t)(l * 4 + 2) * DMOD,
         Vc_h + (size_t)l * B_ * LENC * DMOD, DMOD,
         B_ * LENC, DMOD, DMOD, 2);
  }

  // ---- 16 incremental decode steps ----
  for (int t = 0; t < NT; ++t) {
    k_embed<<<B_, DMOD, 0, stream>>>((const float*)d_out, embW_h, emb_b,
                                     xbuf, xh, t);
    for (int l = 0; l < NL; ++l) {
      const _Float16* sW = selfW_h + (size_t)l * 4 * DMOD * DMOD;
      const float*    sb = self_b + (size_t)l * 4 * DMOD;
      float* kc = kcache + (size_t)l * B_ * NT * DMOD;
      float* vc = vcache + (size_t)l * B_ * NT * DMOD;
      // q/k/v projections (new token only); k/v land directly in the cache
      gemm(stream, xh, DMOD, sW + 0 * DMOD * DMOD, DMOD, sb + 0 * DMOD,
           qbuf, DMOD, B_, DMOD, DMOD, 0);
      gemm(stream, xh, DMOD, sW + 1 * DMOD * DMOD, DMOD, sb + 1 * DMOD,
           kc + (size_t)t * DMOD, NT * DMOD, B_, DMOD, DMOD, 0);
      gemm(stream, xh, DMOD, sW + 2 * DMOD * DMOD, DMOD, sb + 2 * DMOD,
           vc + (size_t)t * DMOD, NT * DMOD, B_, DMOD, DMOD, 0);
      k_self_attn<<<dim3(B_, NH), DH, 0, stream>>>(qbuf, kc, vc, obuf, oh, t);
      gemm(stream, oh, DMOD, sW + 3 * DMOD * DMOD, DMOD, sb + 3 * DMOD,
           abuf, DMOD, B_, DMOD, DMOD, 0);
      k_ln<<<B_, DMOD, 0, stream>>>(xbuf, abuf, ln_g + (size_t)(l * 3 + 0) * DMOD,
                                    ln_b + (size_t)(l * 3 + 0) * DMOD, xbuf, xh);
      // cross attention against cached f16 K/V
      const _Float16* cW = crossW_h + (size_t)l * 4 * DMOD * DMOD;
      const float*    cb = cross_b + (size_t)l * 4 * DMOD;
      gemm(stream, xh, DMOD, cW + 0 * DMOD * DMOD, DMOD, cb + 0 * DMOD,
           qbuf, DMOD, B_, DMOD, DMOD, 0);
      k_cross_attn<<<dim3(B_, NH), 256, 0, stream>>>(
          qbuf, Kc_h + (size_t)l * B_ * LENC * DMOD,
          Vc_h + (size_t)l * B_ * LENC * DMOD, padmask, obuf, oh);
      gemm(stream, oh, DMOD, cW + 3 * DMOD * DMOD, DMOD, cb + 3 * DMOD,
           abuf, DMOD, B_, DMOD, DMOD, 0);
      k_ln<<<B_, DMOD, 0, stream>>>(xbuf, abuf, ln_g + (size_t)(l * 3 + 1) * DMOD,
                                    ln_b + (size_t)(l * 3 + 1) * DMOD, xbuf, xh);
      // FFN: relu GEMM (f16 out) -> GEMM
      gemm(stream, xh, DMOD, ffn1_h + (size_t)l * DMOD * DFF, DFF,
           ffn_b1 + (size_t)l * DFF, hbuf_h, DFF, B_, DFF, DMOD, 1 | 2);
      gemm(stream, hbuf_h, DFF, ffn2_h + (size_t)l * DFF * DMOD, DMOD,
           ffn_b2 + (size_t)l * DMOD, abuf, DMOD, B_, DMOD, DFF, 0);
      k_ln<<<B_, DMOD, 0, stream>>>(xbuf, abuf, ln_g + (size_t)(l * 3 + 2) * DMOD,
                                    ln_b + (size_t)(l * 3 + 2) * DMOD, xbuf, xh);
    }
    // output projection writes token t directly into d_out[:, t, :]
    gemm(stream, xh, DMOD, outW_h, TS, out_b,
         (float*)d_out + (size_t)t * TS, NT * TS, B_, TS, DMOD, 0);
  }
}
